// Self_Attn_1d_57191784513795
// MI455X (gfx1250) — matrix operand order repaired and checked by reading, not compiled, not run on hardware
//
#include <hip/hip_runtime.h>
#include <hip/hip_bf16.h>

// ---------------------------------------------------------------------------
// Self-attention 1d (SAGAN style), B=4, C=256, L=4096, CQK=32, gfx1250 wave32.
//
// Math:  q = Wq x + bq   [B,32,L]
//        k = Wk x + bk   [B,32,L]
//        v = Wv x + bv   [B,256,L]
//        E[j,l] = q_j . k_l ;  A = softmax over last axis
//        out[c,l] = sum_j v[c,j] * A[j,l] ;  y = gamma*out + x
// Key identity (softmax axis != contraction axis):
//        out[c,l] = sum_j exp(E[j,l]) * (v[c,j] / s[j]),  s[j] = sum_l exp(E[j,l])
// -> everything is GEMM + one row-sum pass. All GEMMs use bf16 WMMA
//    (v_wmma_f32_16x16x32_bf16), f32 accumulate.
// ---------------------------------------------------------------------------

#define Bn   4
#define Cn   256
#define Ln   4096
#define CQK  32

typedef __attribute__((ext_vector_type(8)))  __bf16 v8bf;
typedef __attribute__((ext_vector_type(16))) __bf16 v16bf;
typedef __attribute__((ext_vector_type(8)))  float  v8f;

static __device__ inline v16bf cat8(v8bf a, v8bf b) {
    return __builtin_shufflevector(a, b, 0,1,2,3,4,5,6,7,8,9,10,11,12,13,14,15);
}
static __device__ inline v8f vzero8() {
    v8f z = {0.f,0.f,0.f,0.f,0.f,0.f,0.f,0.f};
    return z;
}
static __device__ inline v8f wmma_bf16(v16bf a, v16bf b, v8f c) {
    // D = A(16x32 bf16) * B(32x16 bf16) + C(16x16 f32)
    return __builtin_amdgcn_wmma_f32_16x16x32_bf16(false, a, false, b,
                                                   (short)0, c, false, false);
}

// ---------------------------------------------------------------------------
// Kernel 0a: transpose+convert x[B,C,L] f32 -> xt[B,L,C] bf16 (position-major)
// ---------------------------------------------------------------------------
__global__ __launch_bounds__(256) void k_prep_xt(const float* __restrict__ x,
                                                 __bf16* __restrict__ xt) {
    int tid = blockIdx.x * 256 + threadIdx.x;       // over B*L*C
    int c = tid & (Cn - 1);
    int l = (tid >> 8) & (Ln - 1);
    int b = tid >> 20;
    xt[tid] = (__bf16)x[((size_t)(b * Cn + c)) * Ln + l];
}

// ---------------------------------------------------------------------------
// Kernel 0b: stack + convert W -> wb[320,256] bf16 (rows: 0..31 Wq, 32..63 Wk,
// 64..319 Wv)
// ---------------------------------------------------------------------------
__global__ __launch_bounds__(256) void k_prep_w(const float* __restrict__ Wq,
                                                const float* __restrict__ Wk,
                                                const float* __restrict__ Wv,
                                                __bf16* __restrict__ wb) {
    int tid = blockIdx.x * 256 + threadIdx.x;       // over 320*256
    int row = tid >> 8, c = tid & 255;
    float v;
    if (row < 32)       v = Wq[row * 256 + c];
    else if (row < 64)  v = Wk[(row - 32) * 256 + c];
    else                v = Wv[(row - 64) * 256 + c];
    wb[tid] = (__bf16)v;
}

// ---------------------------------------------------------------------------
// Kernel 1: projections via WMMA.
//   D[o,l] = sum_c wb[o,c] * xt[b,l,c] + bias[o]
//   o-tiles 0..1 -> qT[B,L,32], 2..3 -> kT[B,L,32], 4..19 -> vb[B,C,L]
// One wave per (b, o-tile, l-tile); 8 k-steps of 32 over C=256.
// ---------------------------------------------------------------------------
__global__ __launch_bounds__(256) void k_proj(const __bf16* __restrict__ wb,
                                              const __bf16* __restrict__ xt,
                                              const float* __restrict__ bq,
                                              const float* __restrict__ bk,
                                              const float* __restrict__ bv,
                                              __bf16* __restrict__ qT,
                                              __bf16* __restrict__ kT,
                                              __bf16* __restrict__ vb) {
    int tid  = threadIdx.x;
    int gw   = blockIdx.x * 8 + (tid >> 5);         // 0 .. 20479
    int lane = tid & 31, ln = lane & 15, lh = lane >> 4;
    int b    = gw / (20 * 256);
    int rem  = gw % (20 * 256);
    int ot   = rem >> 8;                            // 0..19
    int l0   = (rem & 255) << 4;
    int o0   = ot << 4;

    const __bf16* xrow = xt + ((size_t)(b * Ln + l0 + ln)) * Cn;
    v8f acc = vzero8();
    #pragma unroll
    for (int c0 = 0; c0 < Cn; c0 += 32) {
        // A (wb rows o0..o0+15): lane lh=0 gets K {0-7,16-23}, lh=1 {8-15,24-31}
        const __bf16* ap = wb + (o0 + ln) * Cn + c0 + 8 * lh;
        v16bf A = cat8(*(const v8bf*)ap, *(const v8bf*)(ap + 16));
        // B (xt cols l0..l0+15): lane lh=0 gets K {0-7,16-23}, lh=1 {8-15,24-31}
        const __bf16* bp = xrow + c0 + 8 * lh;
        v16bf Bm = cat8(*(const v8bf*)bp, *(const v8bf*)(bp + 16));
        acc = wmma_bf16(A, Bm, acc);
    }

    const float* bias;
    int obase;
    if (ot < 2)      { bias = bq; obase = o0; }
    else if (ot < 4) { bias = bk; obase = o0 - 32; }
    else             { bias = bv; obase = o0 - 64; }
    #pragma unroll
    for (int r = 0; r < 8; ++r) acc[r] += bias[obase + 8 * lh + r];

    if (ot < 4) {
        // store position-major [B,L,32]; lane's 8 rows are contiguous in o
        __bf16* dst = (ot < 2) ? qT : kT;
        int oo = (ot < 2) ? o0 : (o0 - 32);         // 0 or 16
        v8bf p;
        #pragma unroll
        for (int r = 0; r < 8; ++r) p[r] = (__bf16)acc[r];
        *(v8bf*)(dst + ((size_t)(b * Ln + l0 + ln)) * CQK + oo + 8 * lh) = p;
    } else {
        int cbase = o0 - 64 + 8 * lh;
        #pragma unroll
        for (int r = 0; r < 8; ++r)
            vb[((size_t)(b * Cn + cbase + r)) * Ln + l0 + ln] = (__bf16)acc[r];
    }
}

// ---------------------------------------------------------------------------
// Kernel 2: rs[b,i] = 1 / sum_j exp(q_i . k_j).
// One wave per (b, i-tile of 16); 256 WMMA j-steps; exp+accumulate in VALU;
// cross-lane reduce over the 16-lane column groups.
// ---------------------------------------------------------------------------
__global__ __launch_bounds__(256) void k_rowsum(const __bf16* __restrict__ qT,
                                                const __bf16* __restrict__ kT,
                                                float* __restrict__ rs) {
    int tid  = threadIdx.x;
    int gw   = blockIdx.x * 8 + (tid >> 5);         // 0..1023
    int lane = tid & 31, ln = lane & 15, lh = lane >> 4;
    int b  = gw >> 8;
    int i0 = (gw & 255) << 4;

    const __bf16* ap = qT + ((size_t)(b * Ln + i0 + ln)) * CQK + 8 * lh;
    v16bf A = cat8(*(const v8bf*)ap, *(const v8bf*)(ap + 16));

    float s[8] = {0.f,0.f,0.f,0.f,0.f,0.f,0.f,0.f};
    for (int j0 = 0; j0 < Ln; j0 += 16) {
        const __bf16* bp = kT + ((size_t)(b * Ln + j0 + ln)) * CQK + 8 * lh;
        v16bf Bm = cat8(*(const v8bf*)bp, *(const v8bf*)(bp + 16));
        v8f e = wmma_bf16(A, Bm, vzero8());
        #pragma unroll
        for (int r = 0; r < 8; ++r) s[r] += __expf(e[r]);
    }
    // reduce over the 16 lanes of each half (columns j of the tile)
    #pragma unroll
    for (int r = 0; r < 8; ++r) {
        #pragma unroll
        for (int off = 1; off < 16; off <<= 1)
            s[r] += __shfl_xor(s[r], off, 16);
    }
    if (ln == 0) {
        #pragma unroll
        for (int r = 0; r < 8; ++r)
            rs[(size_t)b * Ln + i0 + 8 * lh + r] = 1.0f / s[r];
    }
}

// ---------------------------------------------------------------------------
// Kernel 3: out[c,l] = gamma * sum_j exp(E[j,l])*rs[j]*v[c,j] + x[c,l]
// Block = 8 waves handles (b, l-tile of 16), all 256 channels.
// Per 256-j round: each wave computes a 32(j)x16(l) exp(E) subtile with 2
// WMMAs, scales by rs, writes bf16 [l][j] tile to LDS (layout conversion via
// LDS round-trip), barrier, then each wave runs 2 c-tiles x 8 j-substeps of
// output WMMA reading B from LDS and A (v) from global.
// ---------------------------------------------------------------------------
__global__ __launch_bounds__(256) void k_attn_out(const __bf16* __restrict__ qT,
                                                  const __bf16* __restrict__ kT,
                                                  const __bf16* __restrict__ vb,
                                                  const float* __restrict__ rs,
                                                  const float* __restrict__ x,
                                                  const float* __restrict__ gamma_p,
                                                  float* __restrict__ out) {
    __shared__ __bf16 lds_e[16 * 256];              // [l_rel][j_rel] bf16, 8 KB
    int tid  = threadIdx.x;
    int w    = tid >> 5;
    int lane = tid & 31, ln = lane & 15, lh = lane >> 4;
    int b  = blockIdx.x >> 8;
    int l0 = (blockIdx.x & 255) << 4;

    // Fixed energy B operand: k columns l0..l0+15
    const __bf16* kp = kT + ((size_t)(b * Ln + l0 + ln)) * CQK + 8 * lh;
    v16bf KB = cat8(*(const v8bf*)kp, *(const v8bf*)(kp + 16));

    v8f acc0 = vzero8(), acc1 = vzero8();
    int c0 = w * 32, c1 = c0 + 16;

    for (int round = 0; round < 16; ++round) {
        int j0 = round * 256;
        // ---- Phase A: exp(E)*rs for j in [j0+32w, j0+32w+32) -> LDS ----
        #pragma unroll
        for (int t = 0; t < 2; ++t) {
            int jA = j0 + 32 * w + 16 * t;
            const __bf16* qp = qT + ((size_t)(b * Ln + jA + ln)) * CQK + 8 * lh;
            v16bf A = cat8(*(const v8bf*)qp, *(const v8bf*)(qp + 16));
            v8f e = wmma_bf16(A, KB, vzero8());
            const float* rp = rs + (size_t)b * Ln + jA + 8 * lh; // rows of D
            v8bf p;
            #pragma unroll
            for (int r = 0; r < 8; ++r)
                p[r] = (__bf16)(__expf(e[r]) * rp[r]);
            // D rows (j) are contiguous per lane -> one 16B LDS store
            *(v8bf*)(lds_e + ln * 256 + 32 * w + 16 * t + 8 * lh) = p;
        }
        __syncthreads();
        // ---- Phase B: accumulate out for c-tiles c0, c1 over 256 j ----
        #pragma unroll
        for (int sub = 0; sub < 8; ++sub) {
            int jj = sub * 32;
            const __bf16* ep = lds_e + ln * 256 + jj + 8 * lh;
            v16bf EB = cat8(*(const v8bf*)ep, *(const v8bf*)(ep + 16));
            const __bf16* vp0 = vb + ((size_t)(b * Cn + c0 + ln)) * Ln
                                   + j0 + jj + 8 * lh;
            v16bf A0 = cat8(*(const v8bf*)vp0, *(const v8bf*)(vp0 + 16));
            acc0 = wmma_bf16(A0, EB, acc0);
            const __bf16* vp1 = vb + ((size_t)(b * Cn + c1 + ln)) * Ln
                                   + j0 + jj + 8 * lh;
            v16bf A1 = cat8(*(const v8bf*)vp1, *(const v8bf*)(vp1 + 16));
            acc1 = wmma_bf16(A1, EB, acc1);
        }
        __syncthreads();
    }

    float g = *gamma_p;
    #pragma unroll
    for (int r = 0; r < 8; ++r) {
        int c = c0 + 8 * lh + r;
        size_t idx = ((size_t)(b * Cn + c)) * Ln + l0 + ln;
        out[idx] = g * acc0[r] + x[idx];
        c = c1 + 8 * lh + r;
        idx = ((size_t)(b * Cn + c)) * Ln + l0 + ln;
        out[idx] = g * acc1[r] + x[idx];
    }
}

// ---------------------------------------------------------------------------
// Workspace layout (bytes):
//   xt  bf16 [B,L,C]   @ 0         : 8,388,608
//   wb  bf16 [320,256] @ 8388608   :   163,840
//   qT  bf16 [B,L,32]  @ 8552448   : 1,048,576
//   kT  bf16 [B,L,32]  @ 9601024   : 1,048,576
//   vb  bf16 [B,C,L]   @ 10649600  : 8,388,608
//   rs  f32  [B,L]     @ 19038208  :    65,536   (total ~18.2 MB)
// ---------------------------------------------------------------------------
extern "C" void kernel_launch(void* const* d_in, const int* in_sizes, int n_in,
                              void* d_out, int out_size, void* d_ws, size_t ws_size,
                              hipStream_t stream) {
    const float* x     = (const float*)d_in[0];
    const float* Wq    = (const float*)d_in[1];
    const float* bq    = (const float*)d_in[2];
    const float* Wk    = (const float*)d_in[3];
    const float* bk    = (const float*)d_in[4];
    const float* Wv    = (const float*)d_in[5];
    const float* bv    = (const float*)d_in[6];
    const float* gamma = (const float*)d_in[7];
    float* out = (float*)d_out;

    char* ws = (char*)d_ws;
    __bf16* xt = (__bf16*)(ws);
    __bf16* wb = (__bf16*)(ws + 8388608);
    __bf16* qT = (__bf16*)(ws + 8552448);
    __bf16* kT = (__bf16*)(ws + 9601024);
    __bf16* vb = (__bf16*)(ws + 10649600);
    float*  rs = (float*)(ws + 19038208);

    k_prep_xt<<<16384, 256, 0, stream>>>(x, xt);
    k_prep_w <<<320,   256, 0, stream>>>(Wq, Wk, Wv, wb);
    k_proj   <<<2560,  256, 0, stream>>>(wb, xt, bq, bk, bv, qT, kT, vb);
    k_rowsum <<<128,   256, 0, stream>>>(qT, kT, rs);
    k_attn_out<<<1024, 256, 0, stream>>>(qT, kT, vb, rs, x, gamma, out);
}
